// DeeperGCN_45578192945709
// MI455X (gfx1250) — matrix operand order repaired and checked by reading, not compiled, not run on hardware
//
#include <hip/hip_runtime.h>
#include <hip/hip_fp16.h>

typedef _Float16 v16h __attribute__((ext_vector_type(16)));
typedef _Float16 v8h  __attribute__((ext_vector_type(8)));
typedef float    v8f  __attribute__((ext_vector_type(8)));

#define HID 128
#define EPS_MSG 1e-7f

// Async global->LDS copy of 16 bytes per lane (CDNA5 path, ASYNCcnt-tracked).
// ldsoff: per-lane LDS byte offset (low 32 bits of a generic pointer to __shared__).
__device__ __forceinline__ void async_copy_b128(unsigned ldsoff, const void* gaddr) {
    asm volatile("global_load_async_to_lds_b128 %0, %1, off"
                 :: "v"(ldsoff), "v"(gaddr) : "memory");
}
__device__ __forceinline__ void async_wait0() {
    asm volatile("s_wait_asynccnt 0x0" ::: "memory");
}

// ---------------------------------------------------------------- utilities
__global__ void fill_f32(float* __restrict__ p, long n, float v) {
    long i = (long)blockIdx.x * blockDim.x + threadIdx.x;
    if (i < n) p[i] = v;
}

// ---------------------------------------------------------------- atom encoder
// h[n][c] = 0.8 * sum_f atom_emb[f][x[n][f]][c]
__global__ void atom_enc(const int* __restrict__ x, const float* __restrict__ emb,
                         float* __restrict__ h) {
    int n = blockIdx.x, c = threadIdx.x;
    float s = 0.f;
#pragma unroll
    for (int f = 0; f < 9; ++f) {
        int idx = x[n * 9 + f];
        s += emb[((size_t)f * 128 + idx) * HID + c];
    }
    h[(size_t)n * HID + c] = 0.8f * s;
}

// ---------------------------------------------------------------- degree / dinv
__global__ void deg_accum(const int* __restrict__ src, float* __restrict__ deg, int E) {
    int e = blockIdx.x * blockDim.x + threadIdx.x;
    if (e < E) atomicAdd(&deg[src[e]], 1.0f);
}
__global__ void dinv_k(const float* __restrict__ deg, float* __restrict__ dinv, int N) {
    int n = blockIdx.x * blockDim.x + threadIdx.x;
    if (n < N) dinv[n] = rsqrtf(deg[n]);
}

// ---------------------------------------------------------------- GCN propagation
__global__ void prop_edge(const float* __restrict__ h, const int* __restrict__ src,
                          const int* __restrict__ dst, const float* __restrict__ dinv,
                          float* __restrict__ ax, long total) {
    long i = (long)blockIdx.x * 256 + threadIdx.x;
    if (i >= total) return;
    int e = (int)(i >> 7), c = (int)(i & 127);
    int s = src[e], d = dst[e];
    float w = dinv[s] * dinv[d];
    atomicAdd(&ax[(size_t)s * HID + c], h[(size_t)d * HID + c] * w);
}
__global__ void prop_fin(float* __restrict__ h, const float* __restrict__ ax,
                         const float* __restrict__ deg, long total) {
    long i = (long)blockIdx.x * 256 + threadIdx.x;
    if (i >= total) return;
    int n = (int)(i >> 7);
    float hv = h[i];
    h[i] = 0.5f * (hv + ax[i] + hv / deg[n]);
}

// ---------------------------------------------------------------- softmax aggregation
// messages m >= EPS_MSG > 0, so integer atomicMax on float bit patterns is exact.
__global__ void msg_max(const float* __restrict__ hin, const float* __restrict__ ea,
                        const int* __restrict__ src, const int* __restrict__ dst,
                        int* __restrict__ mmax, long total) {
    long i = (long)blockIdx.x * 256 + threadIdx.x;
    if (i >= total) return;
    if (i + 65536 < total) __builtin_prefetch(&ea[i + 65536], 0, 1);
    int e = (int)(i >> 7), c = (int)(i & 127);
    float m = fmaxf(hin[(size_t)src[e] * HID + c] + ea[i], 0.f) + EPS_MSG;
    atomicMax(&mmax[(size_t)dst[e] * HID + c], __float_as_int(m));
}
__global__ void msg_sum(const float* __restrict__ hin, const float* __restrict__ ea,
                        const int* __restrict__ src, const int* __restrict__ dst,
                        const float* __restrict__ mmax, float* __restrict__ denom,
                        float* __restrict__ num, long total) {
    long i = (long)blockIdx.x * 256 + threadIdx.x;
    if (i >= total) return;
    if (i + 65536 < total) __builtin_prefetch(&ea[i + 65536], 0, 1);
    int e = (int)(i >> 7), c = (int)(i & 127);
    float m = fmaxf(hin[(size_t)src[e] * HID + c] + ea[i], 0.f) + EPS_MSG;
    int d = dst[e];
    size_t o = (size_t)d * HID + c;
    float ex = __expf(m - mmax[o]);
    atomicAdd(&denom[o], ex);
    atomicAdd(&num[o], m * ex);
}
__global__ void agg_fin(const float* __restrict__ hin, const float* __restrict__ num,
                        const float* __restrict__ denom, _Float16* __restrict__ outh,
                        long total) {
    long i = (long)blockIdx.x * 256 + threadIdx.x;
    if (i >= total) return;
    outh[i] = (_Float16)(hin[i] + num[i] / denom[i]);
}

// ---------------------------------------------------------------- weight prep
// w1 [4][128][256] -> w1t [4][256][128] (f16, col-major panels for contiguous LDS fill)
__global__ void cvt_w1t(const float* __restrict__ w1, _Float16* __restrict__ w1t) {
    long i = (long)blockIdx.x * 256 + threadIdx.x;
    if (i >= 4L * 128 * 256) return;
    int l = (int)(i >> 15), rem = (int)(i & 32767);
    int n = rem >> 7, k = rem & 127;
    w1t[i] = (_Float16)w1[(size_t)l * 32768 + (size_t)k * 256 + n];
}
// w2 [4][256][128] -> w2t [4][128][256]
__global__ void cvt_w2t(const float* __restrict__ w2, _Float16* __restrict__ w2t) {
    long i = (long)blockIdx.x * 256 + threadIdx.x;
    if (i >= 4L * 128 * 256) return;
    int l = (int)(i >> 15), rem = (int)(i & 32767);
    int n = rem >> 8, k = rem & 255;
    w2t[i] = (_Float16)w2[(size_t)l * 32768 + (size_t)k * 128 + n];
}

// ---------------------------------------------------------------- WMMA GEMM 1
// z[N,256] = A[N,128](f16) @ w1 + b1.  Block: 16 rows x 128 cols, 8 waves, K=128.
union ABfrag { v16h v; v8h h[2]; };

__global__ __launch_bounds__(256) void gemm1_k(const _Float16* __restrict__ A,
                                               const _Float16* __restrict__ Bt, // [256][128] layer slice
                                               const float* __restrict__ bias,  // [256]
                                               float* __restrict__ Z) {
    __shared__ _Float16 As[16 * 128];   //  4 KB
    __shared__ _Float16 Bs[128 * 128];  // 32 KB (cols x K, pre-transposed)
    const int tid = threadIdx.x;
    const int row0 = blockIdx.x * 16;
    const int col0 = blockIdx.y * 128;

    // async LDS fills (rows contiguous in both sources): 16B per lane per issue
    {
        unsigned sa = (unsigned)(size_t)(void*)As + tid * 16u;
        const char* ga = (const char*)(A + (size_t)row0 * 128) + tid * 16;
        async_copy_b128(sa, ga);                       // 16*128*2B = 4 KB
        unsigned sb = (unsigned)(size_t)(void*)Bs + tid * 16u;
        const char* gb = (const char*)(Bt + (size_t)col0 * 128) + tid * 16;
#pragma unroll
        for (int i = 0; i < 8; ++i)                    // 128*128*2B = 32 KB
            async_copy_b128(sb + i * 4096u, gb + i * 4096);
        async_wait0();
    }
    __syncthreads();

    const int wave = tid >> 5, lane = tid & 31;
    const int lm = lane & 15, hi = lane >> 4;
    v8f c = {};
#pragma unroll
    for (int k0 = 0; k0 < 128; k0 += 32) {
        ABfrag a;
        a.h[0] = *(const v8h*)&As[lm * 128 + k0 + 16 * hi];
        a.h[1] = *(const v8h*)&As[lm * 128 + k0 + 16 * hi + 8];
        v16h b = *(const v16h*)&Bs[(wave * 16 + lm) * 128 + k0 + 16 * hi];
        c = __builtin_amdgcn_wmma_f32_16x16x32_f16(false, a.v, false, b,
                                                   (short)0, c, false, false);
    }
    const int col = col0 + wave * 16 + lm;
    const float bv = bias[col];
#pragma unroll
    for (int r = 0; r < 8; ++r)
        Z[(size_t)(row0 + r + 8 * hi) * 256 + col] = c[r] + bv;
}

// ---------------------------------------------------------------- WMMA GEMM 2
// h[N,128] = zr[N,256](f16) @ w2 + b2 (+ residual).  Block: 32 rows x 64 cols, K=256.
__global__ __launch_bounds__(256) void gemm2_k(const _Float16* __restrict__ A,
                                               const _Float16* __restrict__ Bt, // [128][256] layer slice
                                               const float* __restrict__ bias,  // [128]
                                               float* __restrict__ H, int use_res) {
    __shared__ _Float16 As[32 * 256];  // 16 KB
    __shared__ _Float16 Bs[64 * 256];  // 32 KB
    const int tid = threadIdx.x;
    const int row0 = blockIdx.x * 32;
    const int col0 = blockIdx.y * 64;

    {
        unsigned sa = (unsigned)(size_t)(void*)As + tid * 16u;
        const char* ga = (const char*)(A + (size_t)row0 * 256) + tid * 16;
#pragma unroll
        for (int i = 0; i < 4; ++i)                    // 16 KB
            async_copy_b128(sa + i * 4096u, ga + i * 4096);
        unsigned sb = (unsigned)(size_t)(void*)Bs + tid * 16u;
        const char* gb = (const char*)(Bt + (size_t)col0 * 256) + tid * 16;
#pragma unroll
        for (int i = 0; i < 8; ++i)                    // 32 KB
            async_copy_b128(sb + i * 4096u, gb + i * 4096);
        async_wait0();
    }
    __syncthreads();

    const int wave = tid >> 5, lane = tid & 31;
    const int rt = wave >> 2, ct = wave & 3;
    const int lm = lane & 15, hi = lane >> 4;
    v8f c = {};
#pragma unroll
    for (int k0 = 0; k0 < 256; k0 += 32) {
        ABfrag a;
        a.h[0] = *(const v8h*)&As[(rt * 16 + lm) * 256 + k0 + 16 * hi];
        a.h[1] = *(const v8h*)&As[(rt * 16 + lm) * 256 + k0 + 16 * hi + 8];
        v16h b = *(const v16h*)&Bs[(ct * 16 + lm) * 256 + k0 + 16 * hi];
        c = __builtin_amdgcn_wmma_f32_16x16x32_f16(false, a.v, false, b,
                                                   (short)0, c, false, false);
    }
    const int col = col0 + ct * 16 + lm;
    const float bv = bias[col];
#pragma unroll
    for (int r = 0; r < 8; ++r) {
        size_t idx = (size_t)(row0 + rt * 16 + r + 8 * hi) * HID + col;
        float v = c[r] + bv;
        if (use_res) v += H[idx];
        H[idx] = v;
    }
}

// ---------------------------------------------------------------- layernorms
__global__ void ln_relu_256(const float* __restrict__ z, const float* __restrict__ g,
                            const float* __restrict__ b, _Float16* __restrict__ out) {
    int row = blockIdx.x, c = threadIdx.x;  // 256 threads
    float v = z[(size_t)row * 256 + c];
    __shared__ float rs[8], rq[8];
    float s = v, q = v * v;
#pragma unroll
    for (int o = 16; o > 0; o >>= 1) { s += __shfl_down(s, o, 32); q += __shfl_down(q, o, 32); }
    int wv = c >> 5, ln = c & 31;
    if (ln == 0) { rs[wv] = s; rq[wv] = q; }
    __syncthreads();
    float S = 0.f, Q = 0.f;
#pragma unroll
    for (int i = 0; i < 8; ++i) { S += rs[i]; Q += rq[i]; }
    float mu = S * (1.f / 256.f);
    float var = Q * (1.f / 256.f) - mu * mu;
    float o = (v - mu) * rsqrtf(var + 1e-5f) * g[c] + b[c];
    out[(size_t)row * 256 + c] = (_Float16)fmaxf(o, 0.f);
}

// act: 0 none, 1 sigmoid, 2 relu6, 3 relu
__global__ void ln_act_128(const float* __restrict__ in, const float* __restrict__ g,
                           const float* __restrict__ b, float* __restrict__ out, int act) {
    int row = blockIdx.x, c = threadIdx.x;  // 128 threads
    float v = in[(size_t)row * HID + c];
    __shared__ float rs[4], rq[4];
    float s = v, q = v * v;
#pragma unroll
    for (int o = 16; o > 0; o >>= 1) { s += __shfl_down(s, o, 32); q += __shfl_down(q, o, 32); }
    int wv = c >> 5, ln = c & 31;
    if (ln == 0) { rs[wv] = s; rq[wv] = q; }
    __syncthreads();
    float S = rs[0] + rs[1] + rs[2] + rs[3];
    float Q = rq[0] + rq[1] + rq[2] + rq[3];
    float mu = S * (1.f / 128.f);
    float var = Q * (1.f / 128.f) - mu * mu;
    float o = (v - mu) * rsqrtf(var + 1e-5f) * g[c] + b[c];
    if (act == 1) o = 1.f / (1.f + __expf(-o));
    else if (act == 2) o = fminf(fmaxf(o, 0.f), 6.f);
    else if (act == 3) o = fmaxf(o, 0.f);
    out[(size_t)row * HID + c] = o;
}

// ---------------------------------------------------------------- pooling / prediction
__global__ void pool_accum(const float* __restrict__ h, const int* __restrict__ batch,
                           float* __restrict__ hg, float* __restrict__ cnt) {
    int n = blockIdx.x, c = threadIdx.x;
    int g = batch[n];
    atomicAdd(&hg[(size_t)g * HID + c], h[(size_t)n * HID + c]);
    if (c == 0) atomicAdd(&cnt[g], 1.0f);
}
__global__ void predict_k(const float* __restrict__ hg, const float* __restrict__ cnt,
                          const float* __restrict__ pw, const float* __restrict__ pb,
                          const float* __restrict__ y, const float* __restrict__ beta,
                          float* __restrict__ out) {
    int g = blockIdx.x, c = threadIdx.x;  // 128 threads
    float v = (hg[(size_t)g * HID + c] / cnt[g]) * pw[c];
    __shared__ float rs[4];
    float s = v;
#pragma unroll
    for (int o = 16; o > 0; o >>= 1) s += __shfl_down(s, o, 32);
    if ((c & 31) == 0) rs[c >> 5] = s;
    __syncthreads();
    if (c == 0) {
        float d = rs[0] + rs[1] + rs[2] + rs[3] + pb[0];
        float sg = 1.f / (1.f + __expf(-d));
        float be = beta[0];
        out[g] = (1.f - be) * sg + be * y[g * 3 + 2];
    }
}

// ---------------------------------------------------------------- orchestration
extern "C" void kernel_launch(void* const* d_in, const int* in_sizes, int n_in,
                              void* d_out, int out_size, void* d_ws, size_t ws_size,
                              hipStream_t stream) {
    (void)n_in; (void)out_size; (void)ws_size;
    const int*   x        = (const int*)d_in[0];
    const int*   eidx     = (const int*)d_in[1];
    const float* eattr    = (const float*)d_in[2];
    const int*   batch    = (const int*)d_in[3];
    const float* y        = (const float*)d_in[4];
    const float* atom_emb = (const float*)d_in[5];
    const float* gcn_w1   = (const float*)d_in[6];
    const float* gcn_b1   = (const float*)d_in[7];
    const float* gcn_g1   = (const float*)d_in[8];
    const float* gcn_bn1  = (const float*)d_in[9];
    const float* gcn_w2   = (const float*)d_in[10];
    const float* gcn_b2   = (const float*)d_in[11];
    const float* norm_g   = (const float*)d_in[12];
    const float* norm_b   = (const float*)d_in[13];
    const float* pred_w   = (const float*)d_in[14];
    const float* pred_b   = (const float*)d_in[15];
    const float* beta     = (const float*)d_in[16];
    float* out = (float*)d_out;

    const int N = in_sizes[3];          // 51200
    const int E = in_sizes[2] / HID;    // 307200
    const int G = in_sizes[4] / 3;      // 64
    const int* src = eidx;
    const int* dst = eidx + E;

    // ---- workspace layout (aliased) ----
    char* w = (char*)d_ws;
    float* h    = (float*)w;            w += (size_t)N * HID * 4;
    float* hin  = (float*)w;            w += (size_t)N * HID * 4;
    float* Z    = (float*)w;            w += (size_t)N * 256 * 4;  // mmax | denom; later z[N,256]
    float* num  = (float*)w;            w += (size_t)N * HID * 4;  // later zr_h (f16 [N,256])
    _Float16* outh = (_Float16*)w;      w += (size_t)N * HID * 2;
    _Float16* w1t  = (_Float16*)w;      w += 4L * 256 * 128 * 2;
    _Float16* w2t  = (_Float16*)w;      w += 4L * 128 * 256 * 2;
    float* deg  = (float*)w;            w += (size_t)N * 4;
    float* dinv = (float*)w;            w += (size_t)N * 4;
    float* hg   = (float*)w;            w += (size_t)G * HID * 4;
    float* cnt  = (float*)w;            w += (size_t)G * 4;
    int*   mmax  = (int*)Z;
    float* denom = Z + (size_t)N * HID;
    _Float16* zrh = (_Float16*)num;

    const long NC  = (long)N * HID;
    const long EC  = (long)E * HID;
    const unsigned bNC = (unsigned)((NC + 255) / 256);
    const unsigned bEC = (unsigned)((EC + 255) / 256);

    // weights -> transposed f16 panels
    cvt_w1t<<<512, 256, 0, stream>>>(gcn_w1, w1t);
    cvt_w2t<<<512, 256, 0, stream>>>(gcn_w2, w2t);

    // atom encoder
    atom_enc<<<N, HID, 0, stream>>>(x, atom_emb, h);

    // degree / dinv
    fill_f32<<<(N + 255) / 256, 256, 0, stream>>>(deg, N, 1.0f);
    deg_accum<<<(E + 255) / 256, 256, 0, stream>>>(src, deg, E);
    dinv_k<<<(N + 255) / 256, 256, 0, stream>>>(deg, dinv, N);

    // GCN-normalized propagation (ax lives in Z)
    fill_f32<<<bNC, 256, 0, stream>>>(Z, NC, 0.f);
    prop_edge<<<bEC, 256, 0, stream>>>(h, src, dst, dinv, Z, EC);
    prop_fin<<<bNC, 256, 0, stream>>>(h, Z, deg, NC);

    // ---- 4 GENConv layers ----
    for (int l = 0; l < 4; ++l) {
        const float* hinp = h;
        if (l > 0) {
            ln_act_128<<<N, HID, 0, stream>>>(h, norm_g + (l - 1) * HID,
                                              norm_b + (l - 1) * HID, hin, l);
            hinp = hin;
        }
        // softmax aggregation (mmax/denom in Z, num separate)
        fill_f32<<<(unsigned)((2 * NC + 255) / 256), 256, 0, stream>>>(Z, 2 * NC, 0.f);
        fill_f32<<<bNC, 256, 0, stream>>>(num, NC, 0.f);
        msg_max<<<bEC, 256, 0, stream>>>(hinp, eattr, src, dst, mmax, EC);
        msg_sum<<<bEC, 256, 0, stream>>>(hinp, eattr, src, dst, (const float*)mmax,
                                         denom, num, EC);
        agg_fin<<<bNC, 256, 0, stream>>>(hinp, num, denom, outh, NC);

        // MLP: GEMM1 (WMMA) -> LN+ReLU -> GEMM2 (WMMA, +bias +residual)
        gemm1_k<<<dim3(N / 16, 2), 256, 0, stream>>>(outh, w1t + (size_t)l * 32768,
                                                     gcn_b1 + l * 256, Z);
        ln_relu_256<<<N, 256, 0, stream>>>(Z, gcn_g1 + l * 256, gcn_bn1 + l * 256, zrh);
        gemm2_k<<<dim3(N / 32, 2), 256, 0, stream>>>(zrh, w2t + (size_t)l * 32768,
                                                     gcn_b2 + l * HID, h, l > 0 ? 1 : 0);
    }

    // final LN -> hin, mean pool, predict
    ln_act_128<<<N, HID, 0, stream>>>(h, norm_g + 3 * HID, norm_b + 3 * HID, hin, 0);
    fill_f32<<<(unsigned)((G * (HID + 1) + 255) / 256), 256, 0, stream>>>(hg, G * (HID + 1), 0.f);
    pool_accum<<<N, HID, 0, stream>>>(hin, batch, hg, cnt);
    predict_k<<<G, HID, 0, stream>>>(hg, cnt, pred_w, pred_b, y, beta, out);
}
